// LSTMModel_69398081569371
// MI455X (gfx1250) — matrix operand order repaired and checked by reading, not compiled, not run on hardware
//
#include <hip/hip_runtime.h>

typedef _Float16 v16h __attribute__((ext_vector_type(16)));
typedef _Float16 v8h  __attribute__((ext_vector_type(8)));
typedef float    v8f  __attribute__((ext_vector_type(8)));
typedef float    v4f  __attribute__((ext_vector_type(4)));

#define HID    20
#define FEAT   16
#define TSTEPS 512
#define LDG    80   // gates LDS row stride (floats) = 4*HID
#define HPAD   32   // h LDS row stride (halves), K padded 20 -> 32

// CDNA5 has native v_tanh_f32; use it when the toolchain exposes the builtin.
__device__ __forceinline__ float tanh_f(float x) {
#if __has_builtin(__builtin_amdgcn_tanhf)
    return __builtin_amdgcn_tanhf(x);
#else
    float e = __expf(-2.0f * x);
    return (1.0f - e) * __builtin_amdgcn_rcpf(1.0f + e);
#endif
}
__device__ __forceinline__ float sigm_f(float x) {
#if __has_builtin(__builtin_amdgcn_tanhf)
    return fmaf(0.5f, __builtin_amdgcn_tanhf(0.5f * x), 0.5f);
#else
    return __builtin_amdgcn_rcpf(1.0f + __expf(-x));
#endif
}

__global__ __launch_bounds__(32)
void lstm_fused_wmma_kernel(const float* __restrict__ x,
                            const float* __restrict__ w_ih,
                            const float* __restrict__ w_hh,
                            const float* __restrict__ b_ih,
                            const float* __restrict__ b_hh,
                            const float* __restrict__ w_out,
                            const float* __restrict__ b_out,
                            float* __restrict__ out)
{
    __shared__ float     gates_lds[16 * LDG];   // 16 rows x 80 gate cols, f32
    __shared__ _Float16  h_lds[16 * HPAD];      // 16 rows x 32 (20 real + 12 zero), f16

    const int lane = threadIdx.x;   // 0..31 (wave32, one wave per block)
    const int half = lane >> 4;     // 0: lanes 0-15, 1: lanes 16-31
    const int l15  = lane & 15;
    const int b0   = blockIdx.x * 16;

    // ---- zero-init h state in LDS (incl. K-padding 20..31) ----
    #pragma unroll
    for (int k = 0; k < 16; ++k) h_lds[lane * 16 + k] = (_Float16)0.0f;

    // ---- preload weight B-matrices into WMMA B layout (K x N, K padded to 32) ----
    // B layout (16-bit, 32x16): lanes 0-15 hold K{0..7,16..23} of column N=lane,
    //                           lanes 16-31 hold K{8..15,24..31} of column N=lane-16.
    v16h  bIh[5], bHh[5];
    float biasv[5];
    #pragma unroll
    for (int n = 0; n < 5; ++n) {
        const int col = n * 16 + l15;            // gate column 0..79
        v16h bi; v16h bh;
        #pragma unroll
        for (int k = 0; k < 16; ++k) { bi[k] = (_Float16)0.0f; bh[k] = (_Float16)0.0f; }
        const float* wir = w_ih + col * FEAT;    // w_ih: (80,16) row-major
        const float* whr = w_hh + col * HID;     // w_hh: (80,20) row-major
        if (half == 0) {
            #pragma unroll
            for (int k = 0; k < 8; ++k) bi[k] = (_Float16)wir[k];  // K=0..7
            #pragma unroll
            for (int k = 0; k < 8; ++k) bh[k] = (_Float16)whr[k];  // K=0..7
            #pragma unroll
            for (int k = 0; k < 4; ++k)  bh[8 + k] = (_Float16)whr[16 + k]; // K=16..19, rest 0
        } else {
            #pragma unroll
            for (int k = 0; k < 8; ++k) bi[k] = (_Float16)wir[8 + k];  // K=8..15
            #pragma unroll
            for (int k = 0; k < 8; ++k) bh[k] = (_Float16)whr[8 + k];  // K=8..15
        }
        bIh[n] = bi;
        bHh[n] = bh;
        biasv[n] = b_ih[col] + b_hh[col];
    }

    // ---- recurrent state: each lane owns row (lane&15), units {2j + half} ----
    float cst[10];
    #pragma unroll
    for (int j = 0; j < 10; ++j) cst[j] = 0.0f;
    float outacc = 0.0f;

    const int myrow = l15;
    // x: (B, T, F) row-major; this lane streams 8 contiguous floats per step.
    const float* xrow = x + ((size_t)(b0 + myrow) * TSTEPS) * FEAT + half * 8;

    __syncthreads();

    for (int t = 0; t < TSTEPS; ++t) {
        // ---- A(x): 16x32 f16, lane<16 -> K{0..7,16..23}, lane>=16 -> K{8..15,24..31}
        //      real K = 0..15 (F=16), K 16..31 zero-padded.
        //      x is streamed once -> non-temporal loads (don't pollute WGP$/L2).
        v16h aX;
        {
            const v4f* px4 = reinterpret_cast<const v4f*>(xrow + t * FEAT);
            v4f xa = __builtin_nontemporal_load(px4);
            v4f xb = __builtin_nontemporal_load(px4 + 1);
            aX[0] = (_Float16)xa[0]; aX[1] = (_Float16)xa[1];
            aX[2] = (_Float16)xa[2]; aX[3] = (_Float16)xa[3];
            aX[4] = (_Float16)xb[0]; aX[5] = (_Float16)xb[1];
            aX[6] = (_Float16)xb[2]; aX[7] = (_Float16)xb[3];
            #pragma unroll
            for (int k = 8; k < 16; ++k) aX[k] = (_Float16)0.0f;
        }
        // ---- A(h) from LDS (zero-padded K=20..31 already zero in h_lds) ----
        v16h aH;
        {
            const v8h* ph = reinterpret_cast<const v8h*>(h_lds + myrow * HPAD + half * 8);
            v8h lo = ph[0];   // K base .. base+7
            v8h hi = ph[2];   // K base+16 .. base+23
            aH = __builtin_shufflevector(lo, hi, 0,1,2,3,4,5,6,7,8,9,10,11,12,13,14,15);
        }

        // ---- gates = bias + x@w_ih^T + h@w_hh^T  (5 N-tiles, f32 accum) ----
        #pragma unroll
        for (int n = 0; n < 5; ++n) {
            v8f acc;
            #pragma unroll
            for (int j = 0; j < 8; ++j) acc[j] = biasv[n];
            acc = __builtin_amdgcn_wmma_f32_16x16x32_f16(
                      false, aX, false, bIh[n], (short)0, acc, false, false);
            acc = __builtin_amdgcn_wmma_f32_16x16x32_f16(
                      false, aH, false, bHh[n], (short)0, acc, false, false);
            // C/D layout: lane holds column N = l15, rows M = j + half*8
            const int col = n * 16 + l15;
            #pragma unroll
            for (int j = 0; j < 8; ++j)
                gates_lds[(j + half * 8) * LDG + col] = acc[j];
        }
        __syncthreads();

        // ---- elementwise LSTM cell + fused output projection ----
        const float* wo = w_out + t * HID;     // w_out flat index = t*H + u
        const float* gr = gates_lds + myrow * LDG;
        #pragma unroll
        for (int j = 0; j < 10; ++j) {
            const int u = 2 * j + half;        // this lane's hidden unit
            float gi = gr[u];
            float gf = gr[u + HID];
            float gg = gr[u + 2 * HID];
            float go = gr[u + 3 * HID];
            float i_ = sigm_f(gi);
            float f_ = sigm_f(gf);
            float g_ = tanh_f(gg);
            float o_ = sigm_f(go);
            float cn = fmaf(f_, cst[j], i_ * g_);
            cst[j] = cn;
            float h_ = o_ * tanh_f(cn);
            outacc = fmaf(h_, wo[u], outacc);
            h_lds[myrow * HPAD + u] = (_Float16)h_;
        }
        __syncthreads();
    }

    // ---- reduce even/odd-unit partials per row, write output ----
    float tot = outacc + __shfl_xor(outacc, 16, 32);
    if (half == 0) {
        out[b0 + myrow] = tot + b_out[0];
    }
}

extern "C" void kernel_launch(void* const* d_in, const int* in_sizes, int n_in,
                              void* d_out, int out_size, void* d_ws, size_t ws_size,
                              hipStream_t stream) {
    const float* x     = (const float*)d_in[0];
    const float* w_ih  = (const float*)d_in[1];
    const float* w_hh  = (const float*)d_in[2];
    const float* b_ih  = (const float*)d_in[3];
    const float* b_hh  = (const float*)d_in[4];
    const float* w_out = (const float*)d_in[5];
    const float* b_out = (const float*)d_in[6];
    float* out = (float*)d_out;

    const int B = in_sizes[0] / (TSTEPS * FEAT);   // 4096
    dim3 grid(B / 16);                              // one wave per 16-row batch tile
    dim3 block(32);
    lstm_fused_wmma_kernel<<<grid, block, 0, stream>>>(
        x, w_ih, w_hh, b_ih, b_hh, w_out, b_out, out);
}
